// DIFSRwithNSTokens_45414984187926
// MI455X (gfx1250) — compile-verified, matrix operand order checked by reading, not run on hardware
//
#include <hip/hip_runtime.h>
#include <hip/hip_bf16.h>
#include <math.h>

// ---------------------------------------------------------------------------
// Problem constants (from reference): B=32, L=512, H=512, NH=8, HD=64
// ---------------------------------------------------------------------------
#define BB   32
#define LL   512
#define HH   512
#define NHH  8
#define HDD  64
#define MTOT (BB * LL)        // 16384 rows for the projection GEMMs
#define SCALE 0.125f          // HD^-0.5

typedef __attribute__((ext_vector_type(16))) __bf16 v16bf;
typedef __attribute__((ext_vector_type(8)))  __bf16 v8bf;
typedef __attribute__((ext_vector_type(8)))  float  v8f;
typedef __attribute__((ext_vector_type(4)))  float  v4f;

// D = A*B + C, bf16 inputs, f32 accumulate (v_wmma_f32_16x16x32_bf16)
__device__ __forceinline__ v8f wmma_bf16(v16bf a, v16bf b, v8f c) {
  return __builtin_amdgcn_wmma_f32_16x16x32_bf16(
      /*neg_a=*/false, a, /*neg_b=*/false, b,
      /*c_mod=*/(short)0, c, /*reuse_a=*/false, /*reuse_b=*/false);
}

// ---------------------------------------------------------------------------
// Fragment loader. 16-bit A (16x32, MxK) and B (32x16, KxN) share the same
// per-lane mapping: row/col = lane&15 (+row0), elements 0..7 = K = kb+0..7,
// elements 8..15 = K = kb+16..23, with kb = (lane>=16)*8. Both reduce to two
// contiguous 16B loads from a row-major bf16 source with leading dim `ld`.
// ---------------------------------------------------------------------------
__device__ __forceinline__ v16bf load_frag_bf(const __bf16* __restrict__ src,
                                              int row0, int k0, int ld) {
  const int lane = threadIdx.x & 31;
  const __bf16* p = src + (size_t)(row0 + (lane & 15)) * ld + k0 + ((lane >> 4) << 3);
  v8bf lo = *(const v8bf*)p;
  v8bf hi = *(const v8bf*)(p + 16);
  return __builtin_shufflevector(lo, hi, 0,1,2,3,4,5,6,7,8,9,10,11,12,13,14,15);
}

// Cross-lane row reductions: C/D rows live across the 16 lanes of each half
// (lanes 0-15 hold M=v, lanes 16-31 hold M=v+8), so xor masks 1,2,4,8 suffice.
__device__ __forceinline__ float half16_max(float v) {
  v = fmaxf(v, __shfl_xor(v, 1));
  v = fmaxf(v, __shfl_xor(v, 2));
  v = fmaxf(v, __shfl_xor(v, 4));
  v = fmaxf(v, __shfl_xor(v, 8));
  return v;
}
__device__ __forceinline__ float half16_sum(float v) {
  v += __shfl_xor(v, 1);
  v += __shfl_xor(v, 2);
  v += __shfl_xor(v, 4);
  v += __shfl_xor(v, 8);
  return v;
}

// ---------------------------------------------------------------------------
// Kernel 0: elementwise f32 -> bf16 (one-time precision downconvert so the
// GEMM inner loops carry zero converts and half the load bytes).
// Each thread converts 8 contiguous elements (all array sizes divide by 2048).
// ---------------------------------------------------------------------------
__global__ __launch_bounds__(256)
void cvt_bf16_kernel(const float* __restrict__ in, __bf16* __restrict__ out,
                     int n) {
  const int i = (blockIdx.x * 256 + threadIdx.x) * 8;
  if (i >= n) return;
  v4f a = *(const v4f*)(in + i);
  v4f b = *(const v4f*)(in + i + 4);
  v8bf r;
#pragma unroll
  for (int j = 0; j < 4; ++j) { r[j] = (__bf16)a[j]; r[4 + j] = (__bf16)b[j]; }
  *(v8bf*)(out + i) = r;
}

// ---------------------------------------------------------------------------
// Kernel 1: projection GEMM.  out(M=16384, N=512) = X(M,512) * W^T + bias.
// X and W pre-converted to bf16.  W is (N,K) row-major, so B[k][n] = W[n][k]
// -> the B fragment reads row n of W with contiguous k (16B vector loads).
// Block = 256 threads = 8 waves; wave -> 32(M) x 64(N) tile: per K-step
// 8 WMMAs on 6 fragment loads (A-fragment pair shared by all 4 N-tiles).
// Output bf16 head-major (B*NH, L, HD); V transposed to (B*NH, HD, L) so the
// attention P*V B-fragment is contiguous.
// Grid = (N/64, M/256).
// ---------------------------------------------------------------------------
__global__ __launch_bounds__(256)
void proj_gemm_kernel(const __bf16* __restrict__ X, const __bf16* __restrict__ W,
                      const float* __restrict__ bias, __bf16* __restrict__ out,
                      int vtrans) {
  const int wave  = threadIdx.x >> 5;
  const int lane  = threadIdx.x & 31;
  const int mbase = blockIdx.y * 256 + wave * 32;
  const int nbase = blockIdx.x * 64;

  v8f acc0[4] = {}, acc1[4] = {};

#pragma unroll 4
  for (int k0 = 0; k0 < HH; k0 += 32) {
    v16bf a0 = load_frag_bf(X, mbase,      k0, HH);
    v16bf a1 = load_frag_bf(X, mbase + 16, k0, HH);
#pragma unroll
    for (int t = 0; t < 4; ++t) {
      v16bf b = load_frag_bf(W, nbase + t * 16, k0, HH);
      acc0[t] = wmma_bf16(a0, b, acc0[t]);
      acc1[t] = wmma_bf16(a1, b, acc1[t]);
    }
  }

  const int nl   = lane & 15;
  const int half = lane >> 4;
#pragma unroll
  for (int t = 0; t < 4; ++t) {
    const int n  = nbase + t * 16 + nl;
    const float bv = bias[n];
    const int h = n >> 6;        // head
    const int d = n & 63;        // dim within head
#pragma unroll
    for (int mt = 0; mt < 2; ++mt) {
      const v8f* acc = mt ? acc1 : acc0;
#pragma unroll
      for (int v = 0; v < 8; ++v) {
        const int m = mbase + mt * 16 + v + 8 * half;  // global row = b*L + l
        const int bi = m >> 9;                         // batch
        const int l  = m & (LL - 1);                   // seq position
        const float val = acc[t][v] + bv;
        size_t idx;
        if (vtrans) idx = (((size_t)(bi * NHH + h)) * HDD + d) * LL + l;
        else        idx = (((size_t)(bi * NHH + h)) * LL + l) * HDD + d;
        out[idx] = (__bf16)val;
      }
    }
  }
}

// ---------------------------------------------------------------------------
// Kernel 2: fused 3-score flash attention.
// Block = 128 threads = 4 waves; block covers (b,h, 64 q rows), each wave a
// 16-row q tile.  The 6 Q fragments (3 pairs x K=64) are loop-invariant and
// hoisted into registers.  Per 32-col chunk: 12 WMMAs build S, add rel bias,
// causal mask, online softmax, P -> LDS (D-layout -> A-layout transpose),
// 4 WMMAs accumulate O += P*V.  Beyond-diagonal chunks skipped (wave-uniform
// branch, EXEC stays all-ones as WMMA requires).
// Grid = (L/64, B*NH).
// ---------------------------------------------------------------------------
__global__ __launch_bounds__(128)
void attn_kernel(const __bf16* __restrict__ qid, const __bf16* __restrict__ kid,
                 const __bf16* __restrict__ qca, const __bf16* __restrict__ kca,
                 const __bf16* __restrict__ qbr, const __bf16* __restrict__ kbr,
                 const __bf16* __restrict__ vT,  const float*  __restrict__ rel,
                 __bf16* __restrict__ attn) {
  __shared__ __align__(16) __bf16 plds[4][16 * 32];   // 1KB per wave

  const int wave  = threadIdx.x >> 5;
  const int lane  = threadIdx.x & 31;
  const int bh    = blockIdx.y;           // b*NH + h
  const int b     = bh >> 3;
  const int h     = bh & (NHH - 1);
  const int qbase = blockIdx.x * 64 + wave * 16;

  const size_t headoff = (size_t)bh * LL * HDD;
  const __bf16* Q[3] = { qid + headoff, qca + headoff, qbr + headoff };
  const __bf16* K[3] = { kid + headoff, kca + headoff, kbr + headoff };
  const __bf16* vt   = vT + headoff;                       // (HD, L)
  const float*  relp = rel + (size_t)bh * LL * LL;

  const int nl   = lane & 15;
  const int half = lane >> 4;

  // Hoist Q fragments: invariant across the k-chunk loop.
  v16bf qa[6];
#pragma unroll
  for (int p = 0; p < 3; ++p)
#pragma unroll
    for (int ksi = 0; ksi < 2; ++ksi)
      qa[p * 2 + ksi] = load_frag_bf(Q[p], qbase, ksi * 32, HDD);

  float mrow[8], lrow[8];
  v8f o[4] = {};
#pragma unroll
  for (int v = 0; v < 8; ++v) { mrow[v] = -INFINITY; lrow[v] = 0.0f; }

  const int ktmax = (qbase + 15) >> 5;    // last chunk intersecting diagonal
  for (int kt = 0; kt <= ktmax; ++kt) {
    const int kp0 = kt * 32;

    // ---- S = sum of 3 Q_p K_p^T products ----
    v8f s0 = {}, s1 = {};
#pragma unroll
    for (int p = 0; p < 3; ++p) {
#pragma unroll
      for (int ksi = 0; ksi < 2; ++ksi) {
        v16bf b0 = load_frag_bf(K[p], kp0,      ksi * 32, HDD);
        v16bf b1 = load_frag_bf(K[p], kp0 + 16, ksi * 32, HDD);
        s0 = wmma_bf16(qa[p * 2 + ksi], b0, s0);
        s1 = wmma_bf16(qa[p * 2 + ksi], b1, s1);
      }
    }

    // ---- scale + rel bias + causal mask + online softmax update ----
    const int k0g = kp0 + nl;
    const int k1g = k0g + 16;
#pragma unroll
    for (int v = 0; v < 8; ++v) {
      const int q = qbase + v + 8 * half;
      const float* rp = relp + (size_t)q * LL;
      float t0 = s0[v] * SCALE + rp[k0g];
      float t1 = s1[v] * SCALE + rp[k1g];
      if (k0g > q) t0 = -INFINITY;
      if (k1g > q) t1 = -INFINITY;

      const float cmax = half16_max(fmaxf(t0, t1));
      const float nm   = fmaxf(mrow[v], cmax);
      const float alpha = __expf(mrow[v] - nm);   // 0 on first chunk
      mrow[v] = nm;

      const float p0 = __expf(t0 - nm);           // masked -> exp(-inf)=0
      const float p1 = __expf(t1 - nm);
      lrow[v] = lrow[v] * alpha + half16_sum(p0 + p1);
#pragma unroll
      for (int f = 0; f < 4; ++f) o[f][v] *= alpha;

      // stash P (bf16) in D-layout; re-read below in A-layout
      plds[wave][(v + 8 * half) * 32 + nl]      = (__bf16)p0;
      plds[wave][(v + 8 * half) * 32 + 16 + nl] = (__bf16)p1;
    }

    // ---- A-fragment of P from LDS (per-wave slab, DS ops are in-order) ----
    v16bf pa;
    {
      const __bf16* p = &plds[wave][(lane & 15) * 32 + ((lane >> 4) << 3)];
      v8bf lo = *(const v8bf*)p;
      v8bf hi = *(const v8bf*)(p + 16);
      pa = __builtin_shufflevector(lo, hi, 0,1,2,3,4,5,6,7,8,9,10,11,12,13,14,15);
    }

    // ---- O += P * V  (V^T layout -> contiguous 16B B-fragment loads) ----
#pragma unroll
    for (int f = 0; f < 4; ++f) {
      v16bf vb = load_frag_bf(vt, f * 16, kp0, LL);
      o[f] = wmma_bf16(pa, vb, o[f]);
    }
  }

  // ---- normalize and store attn_out (B, L, H) bf16 ----
#pragma unroll
  for (int f = 0; f < 4; ++f) {
#pragma unroll
    for (int v = 0; v < 8; ++v) {
      const int q = qbase + v + 8 * half;
      const float val = o[f][v] / lrow[v];
      attn[((size_t)(b * LL + q)) * HH + h * HDD + f * 16 + nl] = (__bf16)val;
    }
  }
}

// ---------------------------------------------------------------------------
// Kernel 3: output projection. out(16384,512) f32 = attn(bf16) * Wo^T + bo.
// Same 32x64 per-wave tiling as the projection GEMM; f32 coalesced stores.
// ---------------------------------------------------------------------------
__global__ __launch_bounds__(256)
void out_gemm_kernel(const __bf16* __restrict__ A, const __bf16* __restrict__ W,
                     const float* __restrict__ bias, float* __restrict__ out) {
  const int wave  = threadIdx.x >> 5;
  const int lane  = threadIdx.x & 31;
  const int mbase = blockIdx.y * 256 + wave * 32;
  const int nbase = blockIdx.x * 64;

  v8f acc0[4] = {}, acc1[4] = {};
#pragma unroll 4
  for (int k0 = 0; k0 < HH; k0 += 32) {
    v16bf a0 = load_frag_bf(A, mbase,      k0, HH);
    v16bf a1 = load_frag_bf(A, mbase + 16, k0, HH);
#pragma unroll
    for (int t = 0; t < 4; ++t) {
      v16bf b = load_frag_bf(W, nbase + t * 16, k0, HH);
      acc0[t] = wmma_bf16(a0, b, acc0[t]);
      acc1[t] = wmma_bf16(a1, b, acc1[t]);
    }
  }

  const int nl   = lane & 15;
  const int half = lane >> 4;
#pragma unroll
  for (int t = 0; t < 4; ++t) {
    const int n  = nbase + t * 16 + nl;
    const float bv = bias[n];
#pragma unroll
    for (int v = 0; v < 8; ++v) {
      const int m0 = mbase + v + 8 * half;
      out[(size_t)m0 * HH + n]        = acc0[t][v] + bv;
      out[(size_t)(m0 + 16) * HH + n] = acc1[t][v] + bv;
    }
  }
}

// ---------------------------------------------------------------------------
// Host launcher. Workspace layout (bf16 elements; PSZ = B*NH*L*HD = B*L*H):
//   slabs 0..5 : Qid Kid Qcate Kcate Qbrand Kbrand   (16 MB each)
//   slab  6    : V^T          slab 7 : attn_out
//   slabs 8..11: bf16 copies of the 4 f32 activation inputs
//   then       : 8 bf16 weight matrices (512x512 each)
// Total ~196 MB.
// ---------------------------------------------------------------------------
extern "C" void kernel_launch(void* const* d_in, const int* in_sizes, int n_in,
                              void* d_out, int out_size, void* d_ws, size_t ws_size,
                              hipStream_t stream) {
  const float* xin[4] = { (const float*)d_in[0],   // seq_id
                          (const float*)d_in[1],   // side_cate
                          (const float*)d_in[2],   // side_brand
                          (const float*)d_in[3] }; // V_id_input
  const float* rel = (const float*)d_in[4];
  // d_in[5] = attn_mask (causal tril) -> implemented analytically.
  const float* Wf[8] = { (const float*)d_in[6],  (const float*)d_in[8],
                         (const float*)d_in[10], (const float*)d_in[12],
                         (const float*)d_in[14], (const float*)d_in[16],
                         (const float*)d_in[18], (const float*)d_in[20] };
  // order: Wq_id, Wk_id, Wv, Wq_cate, Wk_cate, Wq_brand, Wk_brand, Wo
  const float* bias[8] = { (const float*)d_in[7],  (const float*)d_in[9],
                           (const float*)d_in[11], (const float*)d_in[13],
                           (const float*)d_in[15], (const float*)d_in[17],
                           (const float*)d_in[19], (const float*)d_in[21] };
  float* out = (float*)d_out;

  const size_t PSZ = (size_t)BB * NHH * LL * HDD;   // 8,388,608 bf16 elems
  const size_t WSZ = (size_t)HH * HH;               // 262,144 bf16 elems
  const size_t need = (12 * PSZ + 8 * WSZ) * sizeof(__bf16);
  if (ws_size < need) return;                       // ~196 MB scratch

  __bf16* ws   = (__bf16*)d_ws;
  __bf16* qid  = ws + 0 * PSZ;
  __bf16* kid  = ws + 1 * PSZ;
  __bf16* qca  = ws + 2 * PSZ;
  __bf16* kca  = ws + 3 * PSZ;
  __bf16* qbr  = ws + 4 * PSZ;
  __bf16* kbr  = ws + 5 * PSZ;
  __bf16* vT   = ws + 6 * PSZ;
  __bf16* attn = ws + 7 * PSZ;
  __bf16* xb[4];
  for (int i = 0; i < 4; ++i) xb[i] = ws + (8 + i) * PSZ;
  __bf16* wb[8];
  for (int i = 0; i < 8; ++i) wb[i] = ws + 12 * PSZ + i * WSZ;

  // --- one-time f32 -> bf16 downconverts ---
  for (int i = 0; i < 4; ++i)
    cvt_bf16_kernel<<<(int)(PSZ / 2048), 256, 0, stream>>>(xin[i], xb[i], (int)PSZ);
  for (int i = 0; i < 8; ++i)
    cvt_bf16_kernel<<<(int)(WSZ / 2048), 256, 0, stream>>>(Wf[i], wb[i], (int)WSZ);

  // --- projections (bf16 GEMMs) ---
  dim3 gGemm(HH / 64, MTOT / 256);   // (8, 64)
  dim3 bGemm(256);
  proj_gemm_kernel<<<gGemm, bGemm, 0, stream>>>(xb[0], wb[0], bias[0], qid, 0);
  proj_gemm_kernel<<<gGemm, bGemm, 0, stream>>>(xb[0], wb[1], bias[1], kid, 0);
  proj_gemm_kernel<<<gGemm, bGemm, 0, stream>>>(xb[1], wb[3], bias[3], qca, 0);
  proj_gemm_kernel<<<gGemm, bGemm, 0, stream>>>(xb[1], wb[4], bias[4], kca, 0);
  proj_gemm_kernel<<<gGemm, bGemm, 0, stream>>>(xb[2], wb[5], bias[5], qbr, 0);
  proj_gemm_kernel<<<gGemm, bGemm, 0, stream>>>(xb[2], wb[6], bias[6], kbr, 0);
  proj_gemm_kernel<<<gGemm, bGemm, 0, stream>>>(xb[3], wb[2], bias[2], vT,  1);

  // --- fused 3-score flash attention ---
  dim3 gAttn(LL / 64, BB * NHH);     // (8, 256)
  attn_kernel<<<gAttn, dim3(128), 0, stream>>>(qid, kid, qca, kca, qbr, kbr,
                                               vT, rel, attn);

  // --- output projection ---
  out_gemm_kernel<<<gGemm, bGemm, 0, stream>>>(attn, wb[7], bias[7], out);
}
